// GraphTrans_9594956939313
// MI455X (gfx1250) — compile-verified, matrix operand order checked by reading, not run on hardware
//
#include <hip/hip_runtime.h>

#define NN 50000
#define NE 800000
#define DD 128
#define HEADS 8

typedef __attribute__((ext_vector_type(16))) __bf16 v16bf;
typedef __attribute__((ext_vector_type(8)))  float  v8f;
typedef __attribute__((address_space(3))) unsigned short lds_u16;

__device__ __forceinline__ unsigned short f32_to_bf16(float f) {
    unsigned u = __float_as_uint(f);
    u += 0x7FFFu + ((u >> 16) & 1u);          // round-to-nearest-even
    return (unsigned short)(u >> 16);
}
// order-preserving float<->uint encoding for atomicMax on signed floats
__device__ __forceinline__ unsigned encf(float f) {
    unsigned u = __float_as_uint(f);
    return (u & 0x80000000u) ? ~u : (u | 0x80000000u);
}
__device__ __forceinline__ float decf(unsigned s) {
    unsigned u = (s & 0x80000000u) ? (s & 0x7FFFFFFFu) : ~s;
    return __uint_as_float(u);
}

// CDNA5 async copy: per-lane 16B global -> LDS, tracked by ASYNCcnt
__device__ __forceinline__ void async_ld_b128(unsigned short* lds_dst,
                                              const unsigned short* gsrc) {
    lds_u16* d3 = (lds_u16*)lds_dst;          // addrspace(3): 32-bit LDS offset
    asm volatile("global_load_async_to_lds_b128 %0, %1, off"
                 :: "v"(d3), "v"(gsrc)
                 : "memory");
}
__device__ __forceinline__ void wait_async0() {
    asm volatile("s_wait_asynccnt 0x0" ::: "memory");
}

// ---------------------------------------------------------------------------
// C[M x NB] = A_bf16[M x 128] @ Wt_bf16[NB x 128]^T + bias
// bf16 WMMA, fp32 accumulate. 256 threads = 8 waves (4x2), 128-row tile.
// Tiles pulled into LDS with global_load_async_to_lds_b128.
// ---------------------------------------------------------------------------
template <int NB>
__global__ __launch_bounds__(256) void gemm_wmma(
    const unsigned short* __restrict__ A,    // bf16 [M x 128]
    const unsigned short* __restrict__ Wt,   // bf16 [NB x 128] (pre-transposed)
    const float* __restrict__ bias, float* __restrict__ Cout,
    int M, int ldc, int col_off)
{
    constexpr int LDA = 136;                 // pitch in bf16: 272B -> conflict-free b128
    __shared__ unsigned short sA[128 * LDA];
    __shared__ unsigned short sB[NB * LDA];

    const int t  = threadIdx.x;
    const int m0 = blockIdx.x * 128;

    // A tile: 128 rows x 16 chunks of 8 bf16 (16B per lane per async op)
    for (int idx = t; idx < 128 * 16; idx += 256) {
        int r = idx >> 4, c8 = idx & 15;
        unsigned short* ld = &sA[r * LDA + c8 * 8];
        if (m0 + r < M) {
            async_ld_b128(ld, A + (size_t)(m0 + r) * 128 + c8 * 8);
        } else {
            *(uint4*)ld = make_uint4(0u, 0u, 0u, 0u);   // zero-pad M edge
        }
    }
    // B tile: NB rows x 16 chunks (Wt contiguous bf16)
    for (int idx = t; idx < NB * 16; idx += 256) {
        int n = idx >> 4, c8 = idx & 15;
        async_ld_b128(&sB[n * LDA + c8 * 8], Wt + n * 128 + c8 * 8);
    }
    wait_async0();
    __syncthreads();

    const int lane    = t & 31;
    const int wave    = t >> 5;
    const int wm      = wave & 3;            // 32-row strip
    const int wn      = wave >> 2;           // NB/2-col strip
    constexpr int WN  = NB / 2;
    constexpr int NT  = WN / 16;             // n-frags per wave
    const int colLane = lane & 15;
    const int halfSel = lane >> 4;

    v8f acc[2][NT];
    for (int mi = 0; mi < 2; ++mi)
        for (int ni = 0; ni < NT; ++ni) {
            float bv = bias[wn * WN + ni * 16 + colLane];  // C layout: col = lane%16
            acc[mi][ni] = (v8f){bv, bv, bv, bv, bv, bv, bv, bv};
        }

    union Frag { v16bf v; uint4 q[2]; };

    for (int kk = 0; kk < 128; kk += 32) {
        const int k0 = kk + halfSel * 8;     // ISA 16-bit A/B layout: K chunks k0, k0+16
        Frag af[2];
        for (int mi = 0; mi < 2; ++mi) {
            const unsigned short* p = &sA[(wm * 32 + mi * 16 + colLane) * LDA + k0];
            af[mi].q[0] = *(const uint4*)(p);
            af[mi].q[1] = *(const uint4*)(p + 16);
        }
        Frag bfr[NT];
        for (int ni = 0; ni < NT; ++ni) {
            const unsigned short* p = &sB[(wn * WN + ni * 16 + colLane) * LDA + k0];
            bfr[ni].q[0] = *(const uint4*)(p);
            bfr[ni].q[1] = *(const uint4*)(p + 16);
        }
        for (int mi = 0; mi < 2; ++mi)
            for (int ni = 0; ni < NT; ++ni)
                acc[mi][ni] = __builtin_amdgcn_wmma_f32_16x16x32_bf16(
                    false, af[mi].v, false, bfr[ni].v, (short)0, acc[mi][ni],
                    false, false);
    }

    for (int mi = 0; mi < 2; ++mi)
        for (int ni = 0; ni < NT; ++ni) {
            int col   = col_off + wn * WN + ni * 16 + colLane;
            int rbase = m0 + wm * 32 + mi * 16 + halfSel * 8;
            for (int i = 0; i < 8; ++i) {
                int row = rbase + i;
                if (row < M) Cout[(size_t)row * ldc + col] = acc[mi][ni][i];
            }
        }
}

// ---------------------------------------------------------------------------
// Weight prep: fp32 [128 x ncols] row-major -> bf16 [ncols x 128] (transposed)
// ---------------------------------------------------------------------------
__global__ void prep_w_kernel(const float* __restrict__ W,
                              unsigned short* __restrict__ Wt, int ncols)
{
    int i = blockIdx.x * blockDim.x + threadIdx.x;
    if (i >= 128 * ncols) return;
    int k = i / ncols, n = i % ncols;        // coalesced read
    Wt[n * 128 + k] = f32_to_bf16(W[i]);
}

// ---------------------------------------------------------------------------
// BatchNorm (inference) on x -> h (bf16)
// ---------------------------------------------------------------------------
__global__ void bn_kernel(const float* __restrict__ x,
                          const float* __restrict__ gamma, const float* __restrict__ beta,
                          const float* __restrict__ mean,  const float* __restrict__ var,
                          unsigned short* __restrict__ h)
{
    int i = blockIdx.x * blockDim.x + threadIdx.x;
    if (i >= NN * DD) return;
    int c = i & 127;
    float v = (x[i] - mean[c]) * rsqrtf(var[c] + 1e-5f) * gamma[c] + beta[c];
    h[i] = f32_to_bf16(v);
}

// ---------------------------------------------------------------------------
// Pass 1: score[e,h] = <q[dst],k[src]>_head / 4 ; segment-max via sortable atomicMax
// QKVS layout per node: q[0,128) k[128,256) v[256,384) s[384,512)  (fp32, L2-resident)
// ---------------------------------------------------------------------------
__global__ void edge_score_kernel(const int* __restrict__ src, const int* __restrict__ dst,
                                  const float* __restrict__ QKVS,
                                  float* __restrict__ score, unsigned* __restrict__ menc)
{
    int i = blockIdx.x * blockDim.x + threadIdx.x;
    if (i >= NE * HEADS) return;
    int e = i >> 3, h = i & 7;
    int s = src[e], d = dst[e];
    const float4* q = (const float4*)(QKVS + (size_t)d * 512 + h * 16);
    const float4* k = (const float4*)(QKVS + (size_t)s * 512 + 128 + h * 16);
    float acc = 0.f;
    for (int j = 0; j < 4; ++j) {
        float4 qa = q[j], ka = k[j];
        acc += qa.x * ka.x + qa.y * ka.y + qa.z * ka.z + qa.w * ka.w;
    }
    acc *= 0.25f;                            // 1/sqrt(16)
    score[i] = acc;
    atomicMax(&menc[d * HEADS + h], encf(acc));
}

__global__ void max_fix_kernel(unsigned* __restrict__ m)
{
    int i = blockIdx.x * blockDim.x + threadIdx.x;
    if (i >= NN * HEADS) return;
    unsigned s = m[i];
    ((float*)m)[i] = (s == 0u) ? 0.f : decf(s);   // 0 sentinel == empty segment -> 0
}

// ---------------------------------------------------------------------------
// Pass 2: w = exp(score - m[dst]); denom += w; accum[dst] += w * v[src]
// (normalization deferred: sum(alpha*v) == sum(w*v)/denom)
// ---------------------------------------------------------------------------
__global__ void edge_agg_kernel(const int* __restrict__ src, const int* __restrict__ dst,
                                const float* __restrict__ QKVS, const float* __restrict__ score,
                                const float* __restrict__ m,
                                float* __restrict__ denom, float* __restrict__ accum)
{
    int i = blockIdx.x * blockDim.x + threadIdx.x;
    if (i >= NE * HEADS) return;
    int e = i >> 3, h = i & 7;
    int s = src[e], d = dst[e];
    float w = __expf(score[i] - m[d * HEADS + h]);
    atomicAdd(&denom[d * HEADS + h], w);
    const float* v  = QKVS + (size_t)s * 512 + 256 + h * 16;
    float* out      = accum + (size_t)d * 128 + h * 16;
    for (int c = 0; c < 16; ++c) atomicAdd(&out[c], w * v[c]);
}

// ---------------------------------------------------------------------------
// h_next = relu(accum/denom + skip)  -> bf16
// ---------------------------------------------------------------------------
__global__ void node_fin_kernel(const float* __restrict__ accum, const float* __restrict__ denom,
                                const float* __restrict__ QKVS, unsigned short* __restrict__ hn)
{
    int i = blockIdx.x * blockDim.x + threadIdx.x;
    if (i >= NN * DD) return;
    int n = i >> 7, c = i & 127;
    float den = denom[n * HEADS + (c >> 4)] + 1e-16f;
    float val = accum[i] / den + QKVS[(size_t)n * 512 + 384 + c];
    hn[i] = f32_to_bf16(fmaxf(val, 0.f));
}

// ---------------------------------------------------------------------------
extern "C" void kernel_launch(void* const* d_in, const int* in_sizes, int n_in,
                              void* d_out, int out_size, void* d_ws, size_t ws_size,
                              hipStream_t stream)
{
    // setup_inputs() insertion order:
    // 0:x 1:edge_index 2:batch 3..6:bn{gamma,beta,mean,var}
    // 7+8l: layer l {Wq,bq,Wk,bk,Wv,bv,Ws,bs}   31:out.W 32:out.b
    const float* x     = (const float*)d_in[0];
    const int*   ei    = (const int*)d_in[1];
    const int*   src   = ei;
    const int*   dst   = ei + NE;
    const float* gamma = (const float*)d_in[3];
    const float* beta  = (const float*)d_in[4];
    const float* mean  = (const float*)d_in[5];
    const float* var   = (const float*)d_in[6];

    char* ws = (char*)d_ws;
    auto alloc = [&](size_t bytes) {
        char* p = ws;
        ws += (bytes + 255) & ~(size_t)255;
        return p;
    };
    unsigned short* hA    = (unsigned short*)alloc((size_t)NN * 128 * 2);
    unsigned short* hB    = (unsigned short*)alloc((size_t)NN * 128 * 2);
    float*          QKVS  = (float*)alloc((size_t)NN * 512 * 4);  // 102MB < 192MB L2
    float*          score = (float*)alloc((size_t)NE * 8 * 4);
    unsigned*       menc  = (unsigned*)alloc((size_t)NN * 8 * 4);
    float*          denom = (float*)alloc((size_t)NN * 8 * 4);
    float*          accum = (float*)alloc((size_t)NN * 128 * 4);
    unsigned short* WT    = (unsigned short*)alloc((12 * 16384 + 8192) * 2);

    const dim3 b256(256);
    const int gridNd = (NN * DD + 255) / 256;
    const int gridEH = (NE * HEADS + 255) / 256;
    const int gridNH = (NN * HEADS + 255) / 256;
    const int gGemm  = (NN + 127) / 128;
    const int gW128  = (128 * 128 + 255) / 256;
    const int gW64   = (128 * 64 + 255) / 256;

    // Pre-convert all weights to transposed bf16
    for (int l = 0; l < 3; ++l)
        for (int j = 0; j < 4; ++j)
            prep_w_kernel<<<gW128, b256, 0, stream>>>(
                (const float*)d_in[7 + 8 * l + 2 * j], WT + (l * 4 + j) * 16384, 128);
    prep_w_kernel<<<gW64, b256, 0, stream>>>((const float*)d_in[31], WT + 12 * 16384, 64);

    bn_kernel<<<gridNd, b256, 0, stream>>>(x, gamma, beta, mean, var, hA);

    unsigned short* hcur = hA;
    unsigned short* hnxt = hB;
    for (int l = 0; l < 3; ++l) {
        const float* bq = (const float*)d_in[7 + 8 * l + 1];
        const float* bk = (const float*)d_in[7 + 8 * l + 3];
        const float* bv = (const float*)d_in[7 + 8 * l + 5];
        const float* bs = (const float*)d_in[7 + 8 * l + 7];
        const unsigned short* WtQ = WT + (l * 4 + 0) * 16384;
        const unsigned short* WtK = WT + (l * 4 + 1) * 16384;
        const unsigned short* WtV = WT + (l * 4 + 2) * 16384;
        const unsigned short* WtS = WT + (l * 4 + 3) * 16384;

        gemm_wmma<128><<<gGemm, b256, 0, stream>>>(hcur, WtQ, bq, QKVS, NN, 512, 0);
        gemm_wmma<128><<<gGemm, b256, 0, stream>>>(hcur, WtK, bk, QKVS, NN, 512, 128);
        gemm_wmma<128><<<gGemm, b256, 0, stream>>>(hcur, WtV, bv, QKVS, NN, 512, 256);
        gemm_wmma<128><<<gGemm, b256, 0, stream>>>(hcur, WtS, bs, QKVS, NN, 512, 384);

        hipMemsetAsync(menc, 0, (size_t)NN * 8 * 4, stream);
        hipMemsetAsync(denom, 0, (size_t)NN * 8 * 4, stream);
        hipMemsetAsync(accum, 0, (size_t)NN * 128 * 4, stream);

        edge_score_kernel<<<gridEH, b256, 0, stream>>>(src, dst, QKVS, score, menc);
        max_fix_kernel<<<gridNH, b256, 0, stream>>>(menc);
        edge_agg_kernel<<<gridEH, b256, 0, stream>>>(src, dst, QKVS, score,
                                                     (const float*)menc, denom, accum);
        node_fin_kernel<<<gridNd, b256, 0, stream>>>(accum, denom, QKVS, hnxt);

        unsigned short* t = hcur; hcur = hnxt; hnxt = t;
    }

    const float* bo = (const float*)d_in[32];
    gemm_wmma<64><<<gGemm, b256, 0, stream>>>(hcur, WT + 12 * 16384, bo,
                                              (float*)d_out, NN, 64, 0);

    (void)in_sizes; (void)n_in; (void)out_size; (void)ws_size;
}
